// MPNN_25804163514598
// MI455X (gfx1250) — compile-verified, matrix operand order checked by reading, not run on hardware
//
#include <hip/hip_runtime.h>

#define D_    128
#define NN    5000
#define WIN   10
#define NT    50000
#define NE    131072
#define FWD   45000
#define NET   90000
#define HID_  1024
#define CHUNK 32768

typedef __attribute__((ext_vector_type(16))) __bf16       v16bf;
typedef __attribute__((ext_vector_type(8)))  float        v8f;
typedef __attribute__((ext_vector_type(4)))  unsigned int v4u;

__device__ __forceinline__ unsigned short f2bf(float f) {
  union { float f; unsigned int u; } v; v.f = f;
  unsigned int r = v.u + 0x7FFFu + ((v.u >> 16) & 1u);
  return (unsigned short)(r >> 16);
}
__device__ __forceinline__ float bf2f(unsigned short u) {
  union { unsigned int u; float f; } v; v.u = ((unsigned int)u) << 16; return v.f;
}

// ---- pack weight (KxN f32 row-major) into WMMA B-fragment layout, bf16 ----
// frag f = nt*Kt + kt ; element (kt*32 + kmap(lane,h), nt*16 + (lane&15))
// kmap: kbase=(lane>>4)*8 ; h<8 -> kbase+h ; h>=8 -> kbase+h+8
__global__ void mpnn_pack_b(const float* __restrict__ W, unsigned short* __restrict__ out,
                            int K, int N) {
  int gid = blockIdx.x * 256 + threadIdx.x;
  int Kt = K >> 5;
  int total = (N >> 4) * Kt * 32;
  if (gid >= total) return;
  int f = gid >> 5, lane = gid & 31;
  int nt = f / Kt, kt = f % Kt;
  int laneLo = lane & 15, kb = (lane >> 4) * 8;
  size_t obase = (size_t)f * 512 + lane * 16;
#pragma unroll
  for (int h = 0; h < 16; ++h) {
    int k = kt * 32 + kb + (h & 7) + ((h >> 3) << 4);
    out[obase + h] = f2bf(W[(size_t)k * N + nt * 16 + laneLo]);
  }
}

// ---- generic WMMA GEMM: C(MxN) = act(A(MxK,bf16) @ Bpacked + bias) ----
// Each wave computes NTW 16x16 n-tiles for one m-tile, so one A fragment
// feeds NTW back-to-back WMMAs. grid.y = (N/16)/(4*NTW).
// OUTM: 0 = f32 store, 1 = bf16 store, 2 = f32 atomicAdd scatter via sidx[row]
template<bool RELU, int OUTM, int NTW>
__global__ __launch_bounds__(128) void mpnn_gemm(
    const unsigned short* __restrict__ A, const unsigned short* __restrict__ Bp,
    const float* __restrict__ bias, void* __restrict__ out,
    const int* __restrict__ sidx, int M, int K, int N) {
  const int lane = threadIdx.x & 31;
  const int wv   = threadIdx.x >> 5;
  const int mt   = blockIdx.x;
  const int nt0  = (blockIdx.y * 4 + wv) * NTW;
  const int Kt   = K >> 5;
  const int laneLo = lane & 15;
  const int laneHi = lane >> 4;
  const int arow = mt * 16 + laneLo;
  const size_t bstride = (size_t)Kt * 512;
  const unsigned short* ap = A + (size_t)arow * K + laneHi * 8;
  const unsigned short* bp = Bp + (size_t)nt0 * Kt * 512 + lane * 16;

  v8f acc[NTW];
#pragma unroll
  for (int i = 0; i < NTW; ++i) acc[i] = (v8f)(0.0f);

  if (mt * 16 + 16 <= M) {
    // ---- full tile: clean inner loop, no predication ----
    for (int kt = 0; kt < Kt; ++kt) {
      union { v16bf v; v4u u[2]; } a;
      a.u[0] = *(const v4u*)ap;          // K = kbase .. kbase+7
      a.u[1] = *(const v4u*)(ap + 16);   // K = kbase+16 .. kbase+23
#pragma unroll
      for (int i = 0; i < NTW; ++i) {
        v16bf b = *(const v16bf*)(bp + (size_t)i * bstride);
        acc[i] = __builtin_amdgcn_wmma_f32_16x16x32_bf16(false, a.v, false, b,
                                                         (short)0, acc[i], false, false);
      }
      ap += 32;
      bp += 512;
    }
  } else {
    // ---- boundary tile (readout only): guard A rows ----
    const bool aval = arow < M;
    for (int kt = 0; kt < Kt; ++kt) {
      union { v16bf v; v4u u[2]; } a;
      if (aval) {
        a.u[0] = *(const v4u*)ap;
        a.u[1] = *(const v4u*)(ap + 16);
      } else {
        a.u[0] = (v4u)0u; a.u[1] = (v4u)0u;
      }
#pragma unroll
      for (int i = 0; i < NTW; ++i) {
        v16bf b = *(const v16bf*)(bp + (size_t)i * bstride);
        acc[i] = __builtin_amdgcn_wmma_f32_16x16x32_bf16(false, a.v, false, b,
                                                         (short)0, acc[i], false, false);
      }
      ap += 32;
      bp += 512;
    }
  }

  float bv[NTW];
#pragma unroll
  for (int i = 0; i < NTW; ++i)
    bv[i] = bias ? bias[(nt0 + i) * 16 + laneLo] : 0.0f;

#pragma unroll
  for (int j = 0; j < 8; ++j) {
    int crow = mt * 16 + laneHi * 8 + j;   // C layout: vgpr j -> row j (+8 for hi lanes)
    if (crow >= M) continue;
#pragma unroll
    for (int i = 0; i < NTW; ++i) {
      int col = (nt0 + i) * 16 + laneLo;
      float v = acc[i][j] + bv[i];
      if (RELU) v = fmaxf(v, 0.0f);
      if (OUTM == 0) {
        ((float*)out)[(size_t)crow * N + col] = v;
      } else if (OUTM == 1) {
        ((unsigned short*)out)[(size_t)crow * N + col] = f2bf(v);
      } else {
        atomicAdd((float*)out + (size_t)sidx[crow] * N + col, v);
      }
    }
  }
}

// ---- edge index construction ----
__global__ void mpnn_build_idx(const int* __restrict__ ie, int* ia, int* ib) {
  int e = blockIdx.x * 256 + threadIdx.x;
  if (e < NE) {
    int t = ie[3 * e];
    ia[e] = t * NN + ie[3 * e + 1];
    ib[e] = t * NN + ie[3 * e + 2];
  }
}
__global__ void mpnn_build_tidx(int* ts, int* td) {
  int e = blockIdx.x * 256 + threadIdx.x;
  if (e < NET) {
    if (e < FWD) { ts[e] = e;            td[e] = e + NN; }
    else         { ts[e] = NN + (e-FWD); td[e] = e - FWD; }
  }
}
__global__ void mpnn_count(const int* __restrict__ ib, int* cnt) {
  int e = blockIdx.x * 256 + threadIdx.x;
  if (e < NE) atomicAdd(&cnt[ib[e]], 1);
}

// ---- h init: h[k] = embed[nodes[k/WIN]] ----
__global__ void mpnn_init_h(const int* __restrict__ nodes, const float* __restrict__ emb,
                            float* __restrict__ h, unsigned short* __restrict__ hb) {
  size_t gid = (size_t)blockIdx.x * 256 + threadIdx.x;
  if (gid >= (size_t)NT * D_) return;
  int row = (int)(gid >> 7), c = (int)(gid & 127);
  float v = emb[(size_t)nodes[row / WIN] * D_ + c];
  h[gid] = v;
  hb[gid] = f2bf(v);
}

// ---- gather concat: X[r, 0:128]=h[ia[e]], X[r, 128:256]=h[ib[e]] (bf16) ----
__global__ void mpnn_gather(const float* __restrict__ h, const int* __restrict__ ia,
                            const int* __restrict__ ib, unsigned short* __restrict__ X,
                            int rows, int e0) {
  size_t gid = (size_t)blockIdx.x * 256 + threadIdx.x;
  if (gid >= (size_t)rows * 256) return;
  int r = (int)(gid >> 8), c = (int)(gid & 255);
  int e = e0 + r;
  int src = (c < D_) ? ia[e] : ib[e];
  X[gid] = f2bf(h[(size_t)src * D_ + (c & 127)]);
}

// ---- segment mean normalize -> bf16 ----
__global__ void mpnn_seg_div(const float* __restrict__ agg, const int* __restrict__ cnt,
                             unsigned short* __restrict__ aggb, int temporal) {
  size_t gid = (size_t)blockIdx.x * 256 + threadIdx.x;
  if (gid >= (size_t)NT * D_) return;
  int row = (int)(gid >> 7);
  float c;
  if (temporal) c = (row < NN || row >= NT - NN) ? 1.0f : 2.0f;
  else { int ci = cnt[row]; c = (float)ci; }
  float r = (c > 0.0f) ? agg[gid] / c : 0.0f;
  aggb[gid] = f2bf(r);
}

// ---- GRU combine: z,r,hcand from xg/hg gates (order z,r,h) ----
__global__ void mpnn_gru(const float* __restrict__ xg, const float* __restrict__ hg,
                         float* __restrict__ h, unsigned short* __restrict__ hb) {
  size_t gid = (size_t)blockIdx.x * 256 + threadIdx.x;
  if (gid >= (size_t)NT * D_) return;
  int row = (int)(gid >> 7), c = (int)(gid & 127);
  size_t base = (size_t)row * 384;
  float xz = xg[base + c], xr = xg[base + 128 + c], xh = xg[base + 256 + c];
  float hz = hg[base + c], hr = hg[base + 128 + c], hh = hg[base + 256 + c];
  float z  = 1.0f / (1.0f + __expf(-(xz + hz)));
  float r  = 1.0f / (1.0f + __expf(-(xr + hr)));
  float hc = tanhf(xh + r * hh);
  float hold = h[gid];
  float hn = z * hold + (1.0f - z) * hc;
  h[gid] = hn;
  hb[gid] = f2bf(hn);
}

// ---- readout tail: logits = x(512) @ W3(512x10) + b3, softmax ----
__global__ __launch_bounds__(32) void mpnn_final(const unsigned short* __restrict__ x,
                                                 const float* __restrict__ W3,
                                                 const float* __restrict__ b3,
                                                 float* __restrict__ out) {
  int row = blockIdx.x, lane = threadIdx.x;
  float acc[10];
#pragma unroll
  for (int j = 0; j < 10; ++j) acc[j] = 0.0f;
  for (int k = lane; k < 512; k += 32) {
    float xv = bf2f(x[(size_t)row * 512 + k]);
#pragma unroll
    for (int j = 0; j < 10; ++j) acc[j] += xv * W3[k * 10 + j];
  }
#pragma unroll
  for (int j = 0; j < 10; ++j)
    for (int off = 16; off > 0; off >>= 1) acc[j] += __shfl_xor(acc[j], off, 32);
  if (lane == 0) {
    float lg[10], mx = -1e30f;
#pragma unroll
    for (int j = 0; j < 10; ++j) { lg[j] = acc[j] + b3[j]; mx = fmaxf(mx, lg[j]); }
    float s = 0.0f;
#pragma unroll
    for (int j = 0; j < 10; ++j) { lg[j] = __expf(lg[j] - mx); s += lg[j]; }
#pragma unroll
    for (int j = 0; j < 10; ++j) out[(size_t)row * 10 + j] = lg[j] / s;
  }
}

static inline char* carve(char*& p, size_t bytes) {
  char* r = p;
  p += (bytes + 255) & ~(size_t)255;
  return r;
}

extern "C" void kernel_launch(void* const* d_in, const int* in_sizes, int n_in,
                              void* d_out, int out_size, void* d_ws, size_t ws_size,
                              hipStream_t stream) {
  const int*   int_edges = (const int*)d_in[0];
  const int*   nodes     = (const int*)d_in[1];
  const float* embed     = (const float*)d_in[2];
  const float* msgW1     = (const float*)d_in[3];
  const float* msgb1     = (const float*)d_in[4];
  const float* msgW2     = (const float*)d_in[5];
  const float* msgb2     = (const float*)d_in[6];
  const float* giW       = (const float*)d_in[7];
  const float* giU       = (const float*)d_in[8];
  const float* gib       = (const float*)d_in[9];
  const float* gtW       = (const float*)d_in[10];
  const float* gtU       = (const float*)d_in[11];
  const float* gtb       = (const float*)d_in[12];
  const float* roW1      = (const float*)d_in[13];
  const float* rob1      = (const float*)d_in[14];
  const float* roW2      = (const float*)d_in[15];
  const float* rob2      = (const float*)d_in[16];
  const float* roW3      = (const float*)d_in[17];
  const float* rob3      = (const float*)d_in[18];
  float* out = (float*)d_out;

  char* p = (char*)d_ws;
  float*          h    = (float*)carve(p, (size_t)NT * D_ * 4);
  unsigned short* hb   = (unsigned short*)carve(p, (size_t)NT * D_ * 2);
  float*          agg  = (float*)carve(p, (size_t)NT * D_ * 4);
  unsigned short* aggb = (unsigned short*)carve(p, (size_t)NT * D_ * 2);
  int*            cnt  = (int*)carve(p, (size_t)NT * 4);
  float*          xg   = (float*)carve(p, (size_t)NT * 384 * 4);
  float*          hgb  = (float*)carve(p, (size_t)NT * 384 * 4);
  int*            ia   = (int*)carve(p, (size_t)NE * 4);
  int*            ib   = (int*)carve(p, (size_t)NE * 4);
  int*            ts   = (int*)carve(p, (size_t)NET * 4);
  int*            td   = (int*)carve(p, (size_t)NET * 4);
  unsigned short* X    = (unsigned short*)carve(p, (size_t)CHUNK * 256 * 2);
  unsigned short* hid  = (unsigned short*)carve(p, (size_t)CHUNK * HID_ * 2);
  unsigned short* W1p  = (unsigned short*)carve(p, (size_t)256 * HID_ * 2);
  unsigned short* W2p  = (unsigned short*)carve(p, (size_t)HID_ * D_ * 2);
  unsigned short* giWp = (unsigned short*)carve(p, (size_t)D_ * 384 * 2);
  unsigned short* giUp = (unsigned short*)carve(p, (size_t)D_ * 384 * 2);
  unsigned short* gtWp = (unsigned short*)carve(p, (size_t)D_ * 384 * 2);
  unsigned short* gtUp = (unsigned short*)carve(p, (size_t)D_ * 384 * 2);
  unsigned short* rW1p = (unsigned short*)carve(p, (size_t)D_ * HID_ * 2);
  unsigned short* rW2p = (unsigned short*)carve(p, (size_t)HID_ * 512 * 2);
  unsigned short* r1   = (unsigned short*)carve(p, (size_t)NN * HID_ * 2);
  unsigned short* r2   = (unsigned short*)carve(p, (size_t)NN * 512 * 2);

  auto pack = [&](const float* W, unsigned short* o, int K, int N) {
    int total = (K >> 5) * (N >> 4) * 32;
    mpnn_pack_b<<<(total + 255) / 256, 256, 0, stream>>>(W, o, K, N);
  };
  pack(msgW1, W1p, 256, HID_);
  pack(msgW2, W2p, HID_, D_);
  pack(giW, giWp, D_, 384);
  pack(giU, giUp, D_, 384);
  pack(gtW, gtWp, D_, 384);
  pack(gtU, gtUp, D_, 384);
  pack(roW1, rW1p, D_, HID_);
  pack(roW2, rW2p, HID_, 512);

  mpnn_build_idx<<<(NE + 255) / 256, 256, 0, stream>>>(int_edges, ia, ib);
  mpnn_build_tidx<<<(NET + 255) / 256, 256, 0, stream>>>(ts, td);
  hipMemsetAsync(cnt, 0, (size_t)NT * 4, stream);
  mpnn_count<<<(NE + 255) / 256, 256, 0, stream>>>(ib, cnt);

  const int NEL = NT * D_;  // 6.4M elements
  mpnn_init_h<<<(NEL + 255) / 256, 256, 0, stream>>>(nodes, embed, h, hb);

  for (int t = 0; t < 2; ++t) {
    // ---- interior message passing ----
    hipMemsetAsync(agg, 0, (size_t)NT * D_ * 4, stream);
    for (int e0 = 0; e0 < NE; e0 += CHUNK) {
      int rows = (NE - e0 < CHUNK) ? (NE - e0) : CHUNK;
      int mt = (rows + 15) / 16;
      mpnn_gather<<<rows, 256, 0, stream>>>(h, ia, ib, X, rows, e0);
      mpnn_gemm<true, 1, 4><<<dim3(mt, 4), 128, 0, stream>>>(X, W1p, msgb1, hid, nullptr, rows, 256, HID_);
      mpnn_gemm<true, 2, 2><<<dim3(mt, 1), 128, 0, stream>>>(hid, W2p, msgb2, agg, ib + e0, rows, HID_, D_);
    }
    mpnn_seg_div<<<(NEL + 255) / 256, 256, 0, stream>>>(agg, cnt, aggb, 0);
    mpnn_gemm<false, 0, 2><<<dim3(3125, 3), 128, 0, stream>>>(aggb, giWp, gib, xg, nullptr, NT, D_, 384);
    mpnn_gemm<false, 0, 2><<<dim3(3125, 3), 128, 0, stream>>>(hb, giUp, gib + 384, hgb, nullptr, NT, D_, 384);
    mpnn_gru<<<(NEL + 255) / 256, 256, 0, stream>>>(xg, hgb, h, hb);

    // ---- temporal message passing ----
    hipMemsetAsync(agg, 0, (size_t)NT * D_ * 4, stream);
    for (int e0 = 0; e0 < NET; e0 += CHUNK) {
      int rows = (NET - e0 < CHUNK) ? (NET - e0) : CHUNK;
      int mt = (rows + 15) / 16;
      mpnn_gather<<<rows, 256, 0, stream>>>(h, ts, td, X, rows, e0);
      mpnn_gemm<true, 1, 4><<<dim3(mt, 4), 128, 0, stream>>>(X, W1p, msgb1, hid, nullptr, rows, 256, HID_);
      mpnn_gemm<true, 2, 2><<<dim3(mt, 1), 128, 0, stream>>>(hid, W2p, msgb2, agg, td + e0, rows, HID_, D_);
    }
    mpnn_seg_div<<<(NEL + 255) / 256, 256, 0, stream>>>(agg, nullptr, aggb, 1);
    mpnn_gemm<false, 0, 2><<<dim3(3125, 3), 128, 0, stream>>>(aggb, gtWp, gtb, xg, nullptr, NT, D_, 384);
    mpnn_gemm<false, 0, 2><<<dim3(3125, 3), 128, 0, stream>>>(hb, gtUp, gtb + 384, hgb, nullptr, NT, D_, 384);
    mpnn_gru<<<(NEL + 255) / 256, 256, 0, stream>>>(xg, hgb, h, hb);
  }

  // ---- readout ----
  mpnn_gemm<true, 1, 4><<<dim3(313, 4), 128, 0, stream>>>(hb, rW1p, rob1, r1, nullptr, NN, D_, HID_);
  mpnn_gemm<true, 1, 4><<<dim3(313, 2), 128, 0, stream>>>(r1, rW2p, rob2, r2, nullptr, NN, HID_, 512);
  mpnn_final<<<NN, 32, 0, stream>>>(r2, roW3, rob3, out);
}
